// sLSTMblock_60163901882427
// MI455X (gfx1250) — matrix operand order repaired and checked by reading, not compiled, not run on hardware
//
#include <hip/hip_runtime.h>

// Problem constants (from reference)
#define BB   8
#define TT   2048
#define DD   512
#define HH   512
#define CATK 1024          // D + H concatenated contraction dim
#define NWG  32            // scan workgroups; each owns 16 H-columns
#define ROWS 64            // gate rows per WG: 4 gates x 16 columns
#define SCAN_THREADS 128   // 4 wave32 waves; one 16-col N-tile per wave

typedef float  v8f    __attribute__((ext_vector_type(8)));
typedef __bf16 bf16x16 __attribute__((ext_vector_type(16)));

union FragU { uint4 u[2]; bf16x16 v; };

// float -> bf16 (round-to-nearest-even), kept as raw ushort storage
__device__ inline unsigned short f2bf(float f) {
  unsigned u = __float_as_uint(f);
  unsigned r = u + 0x7fffu + ((u >> 16) & 1u);
  return (unsigned short)(r >> 16);
}

// A-matrix 16x32 bf16 fragment (ISA 7.12.2): lane<16 -> K=[k0..k0+7],[k0+16..k0+23]
// lane>=16 -> shifted by 8. Row-major base with leading dim CATK.
__device__ inline bf16x16 load_fragA(const unsigned short* base, int k0, int lane) {
  int row = lane & 15;
  int kb  = k0 + ((lane & 16) ? 8 : 0);
  const unsigned short* p = base + row * CATK + kb;
  FragU f;
  f.u[0] = *(const uint4*)(p);        // K: kb..kb+7
  f.u[1] = *(const uint4*)(p + 16);   // K: kb+16..kb+23
  return f.v;
}

// B-matrix 32x16 bf16 fragment: lane holds column N=lane&15,
// K=[k0..k0+7],[k0+16..k0+23] (lanes 0-15), shifted by 8 (lanes 16-31).
// base = row-major B^T (our weight rows), leading dim CATK.
__device__ inline bf16x16 load_fragB(const unsigned short* base, int k0, int lane) {
  int n  = lane & 15;
  int kb = k0 + ((lane & 16) ? 8 : 0);
  const unsigned short* p = base + n * CATK + kb;
  FragU f;
  f.u[0] = *(const uint4*)(p);
  f.u[1] = *(const uint4*)(p + 16);
  return f.v;
}

__device__ inline float fast_sigmoid(float x) {
  return 1.0f / (1.0f + __expf(-x));
}
__device__ inline float fast_tanh(float x) {
  x = fminf(15.0f, fmaxf(-15.0f, x));
  float e = __expf(2.0f * x);
  return (e - 1.0f) / (e + 1.0f);
}

// ---------------- prep kernels ----------------

// Combined [W;U] weights, reordered per-WG / per-gate-slice, bf16.
// Layout: wu[wg][srow][col], srow: 0-15 i,16-31 f,32-47 o,48-63 g for that
// WG's 16 H-columns; col 0..511 = W, 512..1023 = U.
__global__ void slstm_prep_weights(const float* __restrict__ Ww,
                                   const float* __restrict__ Uw,
                                   unsigned short* __restrict__ wu) {
  int i = blockIdx.x * blockDim.x + threadIdx.x;
  if (i >= NWG * ROWS * CATK) return;
  int wg   = i >> 16;            // ROWS*CATK = 65536
  int rem  = i & 65535;
  int srow = rem >> 10;
  int col  = rem & 1023;
  int gate = srow >> 4;
  int hh   = wg * 16 + (srow & 15);
  int srcr = gate * HH + hh;
  float v = (col < DD) ? Ww[srcr * DD + col] : Uw[srcr * HH + (col - DD)];
  wu[i] = f2bf(v);
}

__global__ void slstm_prep_x(const float* __restrict__ x,
                             unsigned short* __restrict__ xb) {
  const int n = BB * TT * DD;
  for (int i = blockIdx.x * blockDim.x + threadIdx.x; i < n;
       i += gridDim.x * blockDim.x)
    xb[i] = f2bf(x[i]);
}

// Combined bias in scan row order, zero h ping-pong buffers, reset barrier.
__global__ void slstm_prep_misc(const float* __restrict__ Wb,
                                const float* __restrict__ Ub,
                                float* __restrict__ biasr,
                                unsigned short* __restrict__ hbuf,
                                unsigned int* __restrict__ bar) {
  int i = blockIdx.x * blockDim.x + threadIdx.x;
  if (i < NWG * ROWS) {
    int wg = i >> 6, srow = i & 63;
    int gate = srow >> 4;
    int hh   = wg * 16 + (srow & 15);
    int src  = gate * HH + hh;
    biasr[i] = Wb[src] + Ub[src];
  }
  if (i < 2 * BB * HH) hbuf[i] = 0;   // h0 = 0 (both ping-pong buffers)
  if (i == 0) bar[0] = 0u;            // reset grid barrier each launch
}

// ---------------- persistent fused scan ----------------

__global__ void __launch_bounds__(SCAN_THREADS, 1)
slstm_scan_kernel(const unsigned short* __restrict__ xb,
                  const unsigned short* __restrict__ wu,
                  const float* __restrict__ biasr,
                  const float* __restrict__ alpha,
                  unsigned short* __restrict__ hbuf,
                  unsigned int* __restrict__ bar,
                  float* __restrict__ out) {
  extern __shared__ char smem[];
  unsigned short* wuL = (unsigned short*)smem;          // [ROWS][CATK] bf16, 128 KB
  unsigned short* cat = wuL + ROWS * CATK;              // [16][CATK]  bf16, 32 KB
  float*          zs  = (float*)(cat + 16 * CATK);      // [ROWS][8]   f32,  2 KB

  const int wg   = blockIdx.x;
  const int tid  = threadIdx.x;
  const int wv   = tid >> 5;     // wave id 0..3  -> N-tile (16 gate rows)
  const int lane = tid & 31;     // wave32 lane

  // Preload this WG's 128 KB weight slice into LDS (streamed from HBM once).
  {
    const uint4* src = (const uint4*)(wu + (size_t)wg * ROWS * CATK);
    uint4* dst = (uint4*)wuL;
    for (int q = tid; q < ROWS * CATK / 8; q += SCAN_THREADS) dst[q] = src[q];
  }
  // Zero the 8 batch-padding rows of the A tile (M=8..15 discarded anyway).
  {
    uint4 z4 = {0u, 0u, 0u, 0u};
    uint4* dst = (uint4*)(cat + 8 * CATK);
    for (int q = tid; q < 8 * CATK / 8; q += SCAN_THREADS) dst[q] = z4;
  }
  __syncthreads();

  // Per-lane bias (constant per N column) folded into accumulator init.
  const float bias_n = biasr[wg * ROWS + wv * 16 + (lane & 15)];

  // Time-invariant B fragments: cache all 32 K-steps in VGPRs (256 VGPRs/lane).
  bf16x16 breg[32];
#pragma unroll
  for (int ks = 0; ks < 32; ++ks)
    breg[ks] = load_fragB(wuL + wv * 16 * CATK, ks * 32, lane);

  // Element-wise ownership: thread = b*16 + col. Cell state stays in a register.
  const int   eb   = tid >> 4;          // batch 0..7
  const int   ec   = tid & 15;          // column within this WG's slice
  const int   hcol = wg * 16 + ec;      // global H column
  const float al   = alpha[hcol];
  float c = 0.0f;

  for (int t = 0; t < TT; ++t) {
    // Stage cat = [x_t | h_{t-1}] (8 rows x 1024 bf16 = 16 KB) into LDS.
    {
      const unsigned short* hb = hbuf + (t & 1) * (BB * HH);
      for (int q = tid; q < 8 * CATK / 8; q += SCAN_THREADS) {
        int b  = q >> 7;
        int c8 = (q & 127) * 8;
        const uint4* s = (c8 < DD)
            ? (const uint4*)(xb + ((size_t)b * TT + t) * DD + c8)
            : (const uint4*)(hb + b * HH + (c8 - DD));
        *(uint4*)(cat + b * CATK + c8) = *s;
      }
    }
    __syncthreads();

    // z-tile = cat @ WU^T  (M=16 pad, N=16, K=1024 -> 32 x wmma bf16)
    v8f acc;
#pragma unroll
    for (int r = 0; r < 8; ++r) acc[r] = bias_n;
#pragma unroll
    for (int ks = 0; ks < 32; ++ks) {
      bf16x16 a = load_fragA(cat, ks * 32, lane);
      acc = __builtin_amdgcn_wmma_f32_16x16x32_bf16(
          false, a, false, breg[ks], (short)0, acc, false, false);
    }
    // D layout: VGPR r, lanes 0-15 => M=r (batch), N=lane. Keep M=0..7 only.
    if (lane < 16) {
      int zr = wv * 16 + lane;
#pragma unroll
      for (int r = 0; r < 8; ++r) zs[zr * 8 + r] = acc[r];
    }
    __syncthreads();

    // Gates + state update (bias already folded into z).
    float zi = zs[(0 * 16 + ec) * 8 + eb];
    float zf = zs[(1 * 16 + ec) * 8 + eb];
    float zo = zs[(2 * 16 + ec) * 8 + eb];
    float zg = zs[(3 * 16 + ec) * 8 + eb];
    float ig = fast_sigmoid(zi);
    float fg = fast_sigmoid(zf);
    float og = fast_sigmoid(zo);
    float gg = fast_tanh(zg);
    c = al * (fg * c + ig * gg);
    float h = og * fast_tanh(c);

    out[((size_t)eb * TT + t) * HH + hcol] = h;                  // outs[B,T,H]
    hbuf[((t + 1) & 1) * (BB * HH) + eb * HH + hcol] = f2bf(h);  // next-step h
    if (t == TT - 1) {
      out[(size_t)BB * TT * HH + eb * HH + hcol] = h;            // final h
      out[(size_t)BB * TT * HH + BB * HH + eb * HH + hcol] = c;  // final c
    }

    // Device-wide step barrier (32 co-resident WGs; monotonic counter).
    __threadfence();
    __syncthreads();
    if (tid == 0) {
      __hip_atomic_fetch_add(bar, 1u, __ATOMIC_RELEASE, __HIP_MEMORY_SCOPE_AGENT);
      unsigned int target = (unsigned int)(t + 1) * NWG;
      while (__hip_atomic_load(bar, __ATOMIC_ACQUIRE,
                               __HIP_MEMORY_SCOPE_AGENT) < target)
        __builtin_amdgcn_s_sleep(2);
    }
    __syncthreads();
  }
}

// ---------------- launcher ----------------

extern "C" void kernel_launch(void* const* d_in, const int* in_sizes, int n_in,
                              void* d_out, int out_size, void* d_ws, size_t ws_size,
                              hipStream_t stream) {
  (void)in_sizes; (void)n_in; (void)out_size; (void)ws_size;
  const float* x  = (const float*)d_in[0];
  const float* Ww = (const float*)d_in[1];
  const float* Wb = (const float*)d_in[2];
  const float* Uw = (const float*)d_in[3];
  const float* Ub = (const float*)d_in[4];
  const float* al = (const float*)d_in[5];

  // Workspace layout (~21 MB total)
  char* ws = (char*)d_ws;
  unsigned short* wu    = (unsigned short*)(ws);                    //  4 MB bf16 [W;U] reordered
  unsigned short* xbf   = (unsigned short*)(ws + 4194304);          // 16 MB bf16 x
  float*          biasr = (float*)(ws + 4194304 + 16777216);        //  8 KB combined bias
  unsigned short* hbuf  = (unsigned short*)(ws + 20979712);         // 16 KB h ping-pong (bf16)
  unsigned int*   bar   = (unsigned int*)(ws + 20996096);           // grid barrier counter

  constexpr int SMEM_BYTES = ROWS * CATK * 2 + 16 * CATK * 2 + ROWS * 8 * 4; // 165,888 B

  (void)hipFuncSetAttribute((const void*)slstm_scan_kernel,
                            hipFuncAttributeMaxDynamicSharedMemorySize,
                            SMEM_BYTES);

  slstm_prep_weights<<<(NWG * ROWS * CATK + 255) / 256, 256, 0, stream>>>(Ww, Uw, wu);
  slstm_prep_x<<<4096, 256, 0, stream>>>(x, xbf);
  slstm_prep_misc<<<(2 * BB * HH + 255) / 256, 256, 0, stream>>>(Wb, Ub, biasr, hbuf, bar);
  slstm_scan_kernel<<<NWG, SCAN_THREADS, SMEM_BYTES, stream>>>(
      xbf, wu, biasr, al, hbuf, bar, (float*)d_out);
}
